// SNN_CNN_Hybrid_35296041238885
// MI455X (gfx1250) — matrix operand order repaired and checked by reading, not compiled, not run on hardware
//
#include <hip/hip_runtime.h>
#include <cmath>

typedef _Float16 f16;
typedef __attribute__((ext_vector_type(16))) _Float16 v16h;
typedef __attribute__((ext_vector_type(8)))  float    v8f;

#define DECAY 0.5f
#define THRESH 1.0f

// ---- problem constants ----
// x_seq: (T=16, B=16, 1, 100, 768)
// L1: 1->16ch 5x5 s2 p2  -> (B,16,50,384)
// L2: 16->32ch 3x3 s2 p1 -> (B,32,25,192)
// L3: 32->64ch 3x3 s1 p1 (relu), L4: 64->1 3x3 s1 p1
// bilinear x4 -> (B,1,100,768), softplus
static constexpr int T_ = 16, B_ = 16;
static constexpr int H1 = 50, W1 = 384, C1 = 16;
static constexpr int H2 = 25, W2 = 192, C2 = 32;
static constexpr int C3 = 64;

// halo-padded activation geometries (1-px border, zeroed => unconditional loads)
static constexpr int PH1 = H1 + 2, PW1 = W1 + 2;   // 52 x 386  (s1, f16 NHWC)
static constexpr int PH2 = H2 + 2, PW2 = W2 + 2;   // 27 x 194  (m2h / h3, f16 NHWC)

static constexpr size_t MEM1_N = (size_t)B_*H1*W1*C1;      // NHWC f32 (no halo)
static constexpr size_t S1P_N  = (size_t)B_*PH1*PW1*C1;    // NHWC f16 padded
static constexpr size_t MEM2_N = (size_t)B_*C2*H2*W2;      // NCHW f32
static constexpr size_t M2HP_N = (size_t)B_*PH2*PW2*C2;    // NHWC f16 padded
static constexpr size_t H3P_N  = (size_t)B_*PH2*PW2*C3;    // NHWC f16 padded
static constexpr size_t H4_N   = (size_t)B_*H2*W2;         // f32
static constexpr size_t A2_N   = 2*5*32*16;                // packed A frags (f16)
static constexpr size_t A3_N   = 4*9*32*16;
static constexpr size_t A4_N   = 18*32*16;

static __device__ __forceinline__ v8f wmma_acc(v16h a, v16h b, v8f c) {
  return __builtin_amdgcn_wmma_f32_16x16x32_f16(false, a, false, b, (short)0, c, false, false);
}

// ---------------- weight packing into WMMA A-matrix VGPR layout ----------------
// A-layout (16-bit, 16x32): lane<16 row=lane, halves0..15=K0..15 (contiguous k)
//                           lane>=16 row=lane-16, halves0..15=K16..31
__global__ void pack_weights(const float* __restrict__ w2, const float* __restrict__ w3,
                             const float* __restrict__ w4,
                             f16* __restrict__ a2, f16* __restrict__ a3, f16* __restrict__ a4) {
  const int tid = threadIdx.x;
  // conv2: K = tap*16 + ic (taps=9 -> 5 K-steps of 32, pad tap >= 9 packs ZERO so the
  // corresponding in-bounds-garbage B load is annihilated), 2 oc-tiles of 16
  for (int i = tid; i < (int)A2_N; i += 256) {
    int h = i & 15, lane = (i >> 4) & 31;
    int step = (i >> 9) % 5, tile = (i >> 9) / 5;
    int kl = h + ((lane & 16) ? 16 : 0);
    int k = step * 32 + kl, tap = k >> 4, ic = k & 15;
    int oc = tile * 16 + (lane & 15);
    a2[i] = (f16)((tap < 9) ? w2[(oc * 16 + ic) * 9 + tap] : 0.f);
  }
  // conv3: K = tap*32 + ic (one tap per 32-wide K-step), 4 oc-tiles
  for (int i = tid; i < (int)A3_N; i += 256) {
    int h = i & 15, lane = (i >> 4) & 31;
    int step = (i >> 9) % 9, tile = (i >> 9) / 9;
    int kl = h + ((lane & 16) ? 16 : 0);
    int ic = kl, tap = step;
    int oc = tile * 16 + (lane & 15);
    a3[i] = (f16)w3[(oc * 32 + ic) * 9 + tap];
  }
  // conv4: K = tap*64 + ic (18 K-steps), single real output row (M=0)
  for (int i = tid; i < (int)A4_N; i += 256) {
    int h = i & 15, lane = (i >> 4) & 31, step = i >> 9;
    int kl = h + ((lane & 16) ? 16 : 0);
    int k = step * 32 + kl, tap = k >> 6, ic = k & 63;
    int row = lane & 15;
    a4[i] = (f16)((row == 0 && tap < 9) ? w4[ic * 9 + tap] : 0.f);
  }
}

// ---------------- conv1 + LIF (direct fp32, K=25, memory-bound on x) ----------------
__global__ void conv1_lif(const float* __restrict__ x, const float* __restrict__ w1,
                          const float* __restrict__ b1, float* __restrict__ mem1,
                          f16* __restrict__ s1p, int t) {
  __shared__ float ws[16 * 25];
  __shared__ float bs[16];
  const int tid = threadIdx.x;
  for (int i = tid; i < 400; i += 256) ws[i] = w1[i];
  if (tid < 16) bs[tid] = b1[tid];
  __syncthreads();

  const int gid = blockIdx.x * 256 + tid;           // exactly B*H1*W1 = 307200
  const int ox = gid % W1;
  const int oy = (gid / W1) % H1;
  const int b  = gid / (W1 * H1);
  const float* xb = x + ((size_t)(t * B_ + b) * 100) * 768;

  if (t < 15)  // warm L2 for next timestep's read of this window
    __builtin_prefetch(xb + (size_t)B_ * 100 * 768 + (2 * oy) * 768 + 2 * ox, 0, 1);

  float acc[16];
#pragma unroll
  for (int oc = 0; oc < 16; ++oc) acc[oc] = bs[oc];

  for (int ky = 0; ky < 5; ++ky) {
    int iy = 2 * oy - 2 + ky;
    if ((unsigned)iy >= 100u) continue;
    for (int kx = 0; kx < 5; ++kx) {
      int ix = 2 * ox - 2 + kx;
      if ((unsigned)ix >= 768u) continue;
      float xv = xb[iy * 768 + ix];
      int tap = ky * 5 + kx;
#pragma unroll
      for (int oc = 0; oc < 16; ++oc) acc[oc] = fmaf(xv, ws[oc * 25 + tap], acc[oc]);
    }
  }
  const size_t mi = (((size_t)b * H1 + oy) * W1 + ox) * C1;                 // mem1: unpadded
  const size_t si = (((size_t)b * PH1 + (oy + 1)) * PW1 + (ox + 1)) * C1;   // s1: halo-padded
#pragma unroll
  for (int oc = 0; oc < 16; ++oc) {
    float m = mem1[mi + oc] * DECAY + acc[oc];
    float sp = (m >= THRESH) ? 1.f : 0.f;
    s1p[si + oc] = (f16)sp;
    mem1[mi + oc] = (m >= THRESH) ? 0.f : m;
  }
}

// ---------------- conv2 + LIF (WMMA implicit GEMM: M=32 oc, K=144->160, N=16 px) ----------------
// Padded s1 => unconditional B loads; per-step offset is a cndmask of two inline constants.
__global__ void conv2_lif(const f16* __restrict__ s1p, const f16* __restrict__ a2,
                          const float* __restrict__ b2, float* __restrict__ mem2) {
  const int lane = threadIdx.x & 31;
  const int wv = blockIdx.x * 8 + (threadIdx.x >> 5);   // 600*8 = 4800 tiles
  const int xt = wv % (W2 / 16);
  const int oy = (wv / (W2 / 16)) % H2;
  const int b  = wv / ((W2 / 16) * H2);
  const int x0 = xt * 16;
  const int px = x0 + (lane & 15);

  // base at tap (ky=0,kx=0): input pixel (2*oy-1, 2*px-1) -> padded coords (2*oy, 2*px)
  const f16* lb = s1p + (((size_t)b * PH1 + 2 * oy) * PW1 + 2 * px) * C1;

  // tap sequences: lane<16 -> {0,2,4,6,8}; lane>=16 -> {1,3,5,7,9(pad, A==0)}
  // delta(tap) = (tap/3)*PW1 + (tap%3), in pixels
  constexpr int SW = PW1;
  const int offLo[5] = {0, 2, SW + 1, 2 * SW, 2 * SW + 2};
  const int offHi[5] = {1, SW, SW + 2, 2 * SW + 1, 3 * SW};

  v8f acc0 = {}, acc1 = {};
#pragma unroll
  for (int s = 0; s < 5; ++s) {
    const int d = ((lane < 16) ? offLo[s] : offHi[s]) * C1;
    v16h bf = *(const v16h*)(lb + d);                                        // 32B contiguous
    v16h a0 = *(const v16h*)(a2 + ((size_t)(0 * 5 + s) * 32 + lane) * 16);
    v16h a1 = *(const v16h*)(a2 + ((size_t)(1 * 5 + s) * 32 + lane) * 16);
    acc0 = wmma_acc(a0, bf, acc0);
    acc1 = wmma_acc(a1, bf, acc1);
  }
  // D layout: VGPR r -> M = r + (lane<16 ? 0 : 8), N = lane&15
  const int rbase = (lane >> 4) * 8;
  const size_t pixi = (size_t)oy * W2 + px;
#pragma unroll
  for (int r = 0; r < 8; ++r) {
    int oc0 = rbase + r;
    size_t i0 = ((size_t)b * C2 + oc0) * H2 * W2 + pixi;
    float m0 = mem2[i0] * DECAY + acc0[r] + b2[oc0];
    mem2[i0] = (m0 >= THRESH) ? 0.f : m0;
    int oc1 = oc0 + 16;
    size_t i1 = ((size_t)b * C2 + oc1) * H2 * W2 + pixi;
    float m1 = mem2[i1] * DECAY + acc1[r] + b2[oc1];
    mem2[i1] = (m1 >= THRESH) ? 0.f : m1;
  }
}

// ---------------- mem2 NCHW f32 -> padded NHWC f16 ----------------
__global__ void cvt_mem2(const float* __restrict__ mem2, f16* __restrict__ m2hp) {
  const int gid = blockIdx.x * 256 + threadIdx.x;       // exactly MEM2_N
  int c = gid & 31;
  int t2 = gid >> 5;
  int px = t2 % W2;
  int t3 = t2 / W2;
  int oy = t3 % H2;
  int b  = t3 / H2;
  m2hp[(((size_t)b * PH2 + (oy + 1)) * PW2 + (px + 1)) * C2 + c] =
      (f16)mem2[(((size_t)b * C2 + c) * H2 + oy) * W2 + px];
}

// ---------------- conv3 + relu (WMMA: M=64 oc, K=288, N=16 px) ----------------
// Padded input => each step's B load is base + compile-time immediate offset.
__global__ void conv3_relu(const f16* __restrict__ m2hp, const f16* __restrict__ a3,
                           const float* __restrict__ b3, f16* __restrict__ h3p) {
  const int lane = threadIdx.x & 31;
  const int wv = blockIdx.x * 8 + (threadIdx.x >> 5);
  const int xt = wv % (W2 / 16);
  const int oy = (wv / (W2 / 16)) % H2;
  const int b  = wv / ((W2 / 16) * H2);
  const int px = xt * 16 + (lane & 15);

  // base at tap(0,0): input pixel (oy-1, px-1) -> padded coords (oy, px)
  const f16* lb = m2hp + (((size_t)b * PH2 + oy) * PW2 + px) * C2 + ((lane >> 4) * 16);

  v8f acc[4] = {{}, {}, {}, {}};
#pragma unroll
  for (int step = 0; step < 9; ++step) {
    const int ky = step / 3, kx = step % 3;                 // constants after unroll
    v16h bf = *(const v16h*)(lb + (ky * PW2 + kx) * C2);    // uniform immediate offset
#pragma unroll
    for (int tile = 0; tile < 4; ++tile) {
      v16h a = *(const v16h*)(a3 + ((size_t)(tile * 9 + step) * 32 + lane) * 16);
      acc[tile] = wmma_acc(a, bf, acc[tile]);
    }
  }
  const int rbase = (lane >> 4) * 8;
  const size_t hbase = (((size_t)b * PH2 + (oy + 1)) * PW2 + (px + 1)) * C3;
#pragma unroll
  for (int tile = 0; tile < 4; ++tile) {
#pragma unroll
    for (int r = 0; r < 8; ++r) {
      int oc = tile * 16 + rbase + r;
      float v = acc[tile][r] + b3[oc];
      h3p[hbase + oc] = (f16)(v > 0.f ? v : 0.f);
    }
  }
}

// ---------------- conv4 (WMMA: 1 real oc row, K=576 in 18 steps) ----------------
__global__ void conv4_wmma(const f16* __restrict__ h3p, const f16* __restrict__ a4,
                           const float* __restrict__ b4, float* __restrict__ h4) {
  const int lane = threadIdx.x & 31;
  const int wv = blockIdx.x * 8 + (threadIdx.x >> 5);
  const int xt = wv % (W2 / 16);
  const int oy = (wv / (W2 / 16)) % H2;
  const int b  = wv / ((W2 / 16) * H2);
  const int x0 = xt * 16;
  const int px = x0 + (lane & 15);

  const f16* lb = h3p + (((size_t)b * PH2 + oy) * PW2 + px) * C3 + ((lane >> 4) * 16);

  v8f acc = {};
#pragma unroll
  for (int step = 0; step < 18; ++step) {
    const int tap = step >> 1;
    const int ky = tap / 3, kx = tap % 3;
    v16h bf = *(const v16h*)(lb + (ky * PW2 + kx) * C3 + (step & 1) * 32);  // immediate offset
    v16h a = *(const v16h*)(a4 + ((size_t)step * 32 + lane) * 16);
    acc = wmma_acc(a, bf, acc);
  }
  if (lane < 16)   // M=0 row lives in acc[0] of lanes 0..15
    h4[((size_t)b * H2 + oy) * W2 + x0 + lane] = acc[0] + b4[0];
}

// ---------------- bilinear x4 (align_corners=False) + softplus ----------------
__global__ void upsample_softplus(const float* __restrict__ h4, float* __restrict__ out) {
  const int gid = blockIdx.x * 256 + threadIdx.x;       // exactly B*100*768
  int ox = gid % 768;
  int oy = (gid / 768) % 100;
  int b  = gid / (768 * 100);
  float fy = (oy + 0.5f) * 0.25f - 0.5f;
  float fx = (ox + 0.5f) * 0.25f - 0.5f;
  float y0f = floorf(fy), x0f = floorf(fx);
  float wy = fy - y0f, wx = fx - x0f;
  int y0 = (int)y0f, x0 = (int)x0f;
  int y0c = min(max(y0, 0), H2 - 1), y1c = min(max(y0 + 1, 0), H2 - 1);
  int x0c = min(max(x0, 0), W2 - 1), x1c = min(max(x0 + 1, 0), W2 - 1);
  const float* hb = h4 + (size_t)b * H2 * W2;
  float v00 = hb[y0c * W2 + x0c], v01 = hb[y0c * W2 + x1c];
  float v10 = hb[y1c * W2 + x0c], v11 = hb[y1c * W2 + x1c];
  float v = (1.f - wy) * ((1.f - wx) * v00 + wx * v01) + wy * ((1.f - wx) * v10 + wx * v11);
  out[gid] = (v > 20.f) ? v : log1pf(__expf(v));
}

extern "C" void kernel_launch(void* const* d_in, const int* in_sizes, int n_in,
                              void* d_out, int out_size, void* d_ws, size_t ws_size,
                              hipStream_t stream) {
  (void)in_sizes; (void)n_in; (void)out_size; (void)ws_size;
  const float* x  = (const float*)d_in[0];
  const float* w1 = (const float*)d_in[1];
  const float* b1 = (const float*)d_in[2];
  const float* w2 = (const float*)d_in[3];
  const float* b2 = (const float*)d_in[4];
  const float* w3 = (const float*)d_in[5];
  const float* b3 = (const float*)d_in[6];
  const float* w4 = (const float*)d_in[7];
  const float* b4 = (const float*)d_in[8];

  char* base = (char*)d_ws;
  size_t off = 0;
  auto take = [&](size_t bytes) -> char* {
    char* p = base + off;
    off = (off + bytes + 255) & ~(size_t)255;
    return p;
  };
  float* mem1 = (float*)take(MEM1_N * 4);
  f16*   s1p  = (f16*)  take(S1P_N * 2);
  float* mem2 = (float*)take(MEM2_N * 4);
  f16*   m2hp = (f16*)  take(M2HP_N * 2);
  f16*   h3p  = (f16*)  take(H3P_N * 2);
  float* h4   = (float*)take(H4_N * 4);
  f16*   a2   = (f16*)  take(A2_N * 2);
  f16*   a3   = (f16*)  take(A3_N * 2);
  f16*   a4   = (f16*)  take(A4_N * 2);

  hipMemsetAsync(mem1, 0, MEM1_N * 4, stream);
  hipMemsetAsync(mem2, 0, MEM2_N * 4, stream);
  hipMemsetAsync(s1p,  0, S1P_N * 2, stream);   // zero halos (interior overwritten each step)
  hipMemsetAsync(m2hp, 0, M2HP_N * 2, stream);
  hipMemsetAsync(h3p,  0, H3P_N * 2, stream);
  pack_weights<<<1, 256, 0, stream>>>(w2, w3, w4, a2, a3, a4);

  for (int t = 0; t < T_; ++t) {
    conv1_lif<<<1200, 256, 0, stream>>>(x, w1, b1, mem1, s1p, t);  // B*H1*W1 threads
    conv2_lif<<<600, 256, 0, stream>>>(s1p, a2, b2, mem2);         // 4800 waves
  }
  cvt_mem2<<<9600, 256, 0, stream>>>(mem2, m2hp);
  conv3_relu<<<600, 256, 0, stream>>>(m2hp, a3, b3, h3p);
  conv4_wmma<<<600, 256, 0, stream>>>(h3p, a4, b4, h4);
  upsample_softplus<<<4800, 256, 0, stream>>>(h4, (float*)d_out);
}
